// GATLayer_85229331022399
// MI455X (gfx1250) — matrix operand order repaired and checked by reading, not compiled, not run on hardware
//
#include <hip/hip_runtime.h>
#include <hip/hip_bf16.h>
#include <hip/hip_fp16.h>

typedef __attribute__((ext_vector_type(16))) _Float16 v16h;
typedef __attribute__((ext_vector_type(8)))  float    v8f;

#define NN      4096
#define IN_DIM  128
#define OUT_DIM 64
#define NHEAD   4
#define ALPHA   0.2f

union V16HU { uint4 q[2]; v16h h; };

// ---------------------------------------------------------------------------
// Kernel 0: pre-swizzle W into the per-lane WMMA B-operand layout (f16).
// Wp[hd][kb][ot][lane][0..15]:  lane<16 -> N=l16, K=kb*32+{0..7,16..23}
//                               lane>=16 -> N=l16, K=kb*32+{8..15,24..31}
// 2048 threads total; each emits one lane's 16 contiguous halves (32 B).
// ---------------------------------------------------------------------------
__global__ __launch_bounds__(256) void gat_wprep(const float* __restrict__ W,
                                                 _Float16* __restrict__ Wp) {
  const int t    = blockIdx.x * 256 + threadIdx.x;   // 0..2047
  const int lane = t & 31;
  const int ot   = (t >> 5) & 3;
  const int kb   = (t >> 7) & 3;
  const int hd   = (t >> 9) & 3;
  const int half = lane >> 4;
  const int l16  = lane & 15;
  const int ncol = ot * 16 + l16;

  _Float16 tmp[16];
#pragma unroll
  for (int j = 0; j < 16; ++j) {
    const float v = W[((size_t)hd * IN_DIM + kb * 32 + 8 * half + (j & 7) + 16 * (j >> 3)) * OUT_DIM + ncol];
    tmp[j] = (_Float16)v;
  }
  _Float16* dst = Wp + (size_t)t * 16;
  *(uint4*)(dst + 0) = *(const uint4*)(tmp + 0);
  *(uint4*)(dst + 8) = *(const uint4*)(tmp + 8);
}

// ---------------------------------------------------------------------------
// Kernel 1: h = x @ W per head, stored transposed as f16: hT[h][o][n].
// One wave computes a 16(row) x 64(out) tile with v_wmma_f32_16x16x32_f16.
// ---------------------------------------------------------------------------
__global__ __launch_bounds__(32) void gat_hproj(const float* __restrict__ x,
                                                const _Float16* __restrict__ Wp,
                                                _Float16* __restrict__ hT) {
  const int hd   = blockIdx.x >> 8;          // 4 heads
  const int n0   = (blockIdx.x & 255) << 4;  // 256 row tiles of 16
  const int lane = threadIdx.x & 31;
  const int half = lane >> 4;
  const int l16  = lane & 15;
  const int row  = n0 + l16;

  v8f acc[4] = {};

  for (int k0 = 0; k0 < IN_DIM; k0 += 32) {
    // A operand (16x32 f16): lane<16 covers K {0..7,16..23}, lane>=16 {8..15,24..31}
    const float* xb = x + (size_t)row * IN_DIM + k0 + 8 * half;
    float4 a0 = *(const float4*)(xb + 0);
    float4 a1 = *(const float4*)(xb + 4);
    float4 a2 = *(const float4*)(xb + 16);
    float4 a3 = *(const float4*)(xb + 20);
    v16h A;
    A[0]=(_Float16)a0.x;  A[1]=(_Float16)a0.y;  A[2]=(_Float16)a0.z;  A[3]=(_Float16)a0.w;
    A[4]=(_Float16)a1.x;  A[5]=(_Float16)a1.y;  A[6]=(_Float16)a1.z;  A[7]=(_Float16)a1.w;
    A[8]=(_Float16)a2.x;  A[9]=(_Float16)a2.y;  A[10]=(_Float16)a2.z; A[11]=(_Float16)a2.w;
    A[12]=(_Float16)a3.x; A[13]=(_Float16)a3.y; A[14]=(_Float16)a3.z; A[15]=(_Float16)a3.w;

#pragma unroll
    for (int ot = 0; ot < 4; ++ot) {
      const _Float16* bp =
          Wp + (size_t)((((hd * 4) + (k0 >> 5)) * 4 + ot) * 32 + lane) * 16;
      V16HU B;
      B.q[0] = *(const uint4*)(bp + 0);
      B.q[1] = *(const uint4*)(bp + 8);
      acc[ot] = __builtin_amdgcn_wmma_f32_16x16x32_f16(
          false, A, false, B.h, (short)0, acc[ot], false, false);
    }
  }

  // C layout: VGPR r -> lanes 0-15: (M=r, N=l16), lanes 16-31: (M=r+8, N=l16).
  // Pack 8 consecutive n per lane into one 16-byte f16x8 store.
#pragma unroll
  for (int ot = 0; ot < 4; ++ot) {
    _Float16 tmp[8];
#pragma unroll
    for (int r = 0; r < 8; ++r) tmp[r] = (_Float16)acc[ot][r];
    _Float16* dst = hT + (size_t)(hd * OUT_DIM + ot * 16 + l16) * NN + n0 + 8 * half;
    *(uint4*)dst = *(const uint4*)tmp;
  }
}

// ---------------------------------------------------------------------------
// Kernel 2: s_self[h][n], s_neigh[h][n] from transposed hT (coalesced over n)
// ---------------------------------------------------------------------------
__global__ __launch_bounds__(256) void gat_scores(const _Float16* __restrict__ hT,
                                                  const float* __restrict__ a_self,
                                                  const float* __restrict__ a_neigh,
                                                  float* __restrict__ sself,
                                                  float* __restrict__ sneigh) {
  const int hd = blockIdx.x >> 4;
  const int n  = ((blockIdx.x & 15) << 8) + threadIdx.x;
  float ss = 0.f, sn = 0.f;
#pragma unroll 8
  for (int o = 0; o < OUT_DIM; ++o) {
    float v = (float)hT[(size_t)(hd * OUT_DIM + o) * NN + n];
    ss += v * a_self[hd * OUT_DIM + o];
    sn += v * a_neigh[hd * OUT_DIM + o];
  }
  sself[hd * NN + n]  = ss;
  sneigh[hd * NN + n] = sn;
}

// ---------------------------------------------------------------------------
// Kernel 3: flash-style attention. Block = 16-row tile; 4 waves = 4 heads.
// Pass A: exact row max (pure v_max, no exp).
// Pass B: e = exp(s - M) packed to f16 in the WMMA A lane layout; the row sum
//         is accumulated alongside and 1/S applied in the epilogue, so only
//         ONE v_exp_f32 per score element total.
// ---------------------------------------------------------------------------
__global__ __launch_bounds__(128) void gat_attn(const float* __restrict__ adj,
                                                const _Float16* __restrict__ hT,
                                                const float* __restrict__ sself,
                                                const float* __restrict__ sneigh,
                                                float* __restrict__ out) {
  __shared__ float snl[NHEAD][NN];   // 64 KB of the 320 KB WGP LDS

  const int tile = blockIdx.x;          // 256 row tiles
  const int hd   = threadIdx.x >> 5;    // wave id == head
  const int lane = threadIdx.x & 31;
  const int half = lane >> 4;
  const int l16  = lane & 15;
  const int row  = tile * 16 + l16;

  // stage this head's s_neigh row into LDS (coalesced float4)
  {
    const float4* src = (const float4*)(sneigh + (size_t)hd * NN);
    float4* dst = (float4*)snl[hd];
    for (int i = lane; i < NN / 4; i += 32) dst[i] = src[i];
  }
  __syncthreads();

  const float  ss     = sself[hd * NN + row];
  const float* adjrow = adj + (size_t)row * NN;

  // ----- Pass A: exact max over this lane's K subset -----
  float M = -1e30f;
  for (int m0 = 0; m0 < NN; m0 += 32) {
    __builtin_prefetch(adjrow + m0 + 512, 0, 0);
    const int c0 = m0 + 8 * half;
#pragma unroll
    for (int rb = 0; rb < 2; ++rb) {
      const int c = c0 + rb * 16;
      float4 av0 = *(const float4*)(adjrow + c);
      float4 av1 = *(const float4*)(adjrow + c + 4);
      float4 sv0 = *(const float4*)(&snl[hd][c]);
      float4 sv1 = *(const float4*)(&snl[hd][c + 4]);
      float aarr[8] = {av0.x, av0.y, av0.z, av0.w, av1.x, av1.y, av1.z, av1.w};
      float sarr[8] = {sv0.x, sv0.y, sv0.z, sv0.w, sv1.x, sv1.y, sv1.z, sv1.w};
#pragma unroll
      for (int j = 0; j < 8; ++j) {
        float s = ss + sarr[j];
        s = (s > 0.f) ? s : ALPHA * s;
        M = fmaxf(M, (aarr[j] != 0.f) ? s : -1e30f);
      }
    }
  }
  // row max: merge lane pair (l, l+16) which split the row's columns
  M = fmaxf(M, __shfl_xor(M, 16, 32));

  // ----- Pass B: out += exp(S) @ h via WMMA, sum accumulated alongside -----
  v8f acc[4] = {};
  float S = 0.f;
  for (int m0 = 0; m0 < NN; m0 += 32) {
    const int c0 = m0 + 16 * half;  // lane<16: K {0..15}; lane>=16: {16..31}
    v16h A;
#pragma unroll
    for (int rb = 0; rb < 2; ++rb) {
      const int c = c0 + rb * 8;
      float4 av0 = *(const float4*)(adjrow + c);
      float4 av1 = *(const float4*)(adjrow + c + 4);
      float4 sv0 = *(const float4*)(&snl[hd][c]);
      float4 sv1 = *(const float4*)(&snl[hd][c + 4]);
      float aarr[8] = {av0.x, av0.y, av0.z, av0.w, av1.x, av1.y, av1.z, av1.w};
      float sarr[8] = {sv0.x, sv0.y, sv0.z, sv0.w, sv1.x, sv1.y, sv1.z, sv1.w};
#pragma unroll
      for (int j = 0; j < 8; ++j) {
        float s = ss + sarr[j];
        s = (s > 0.f) ? s : ALPHA * s;
        const float e = (aarr[j] != 0.f) ? __expf(s - M) : 0.f;  // e in [0,1]
        S += e;
        A[rb * 8 + j] = (_Float16)e;
      }
    }
#pragma unroll
    for (int ot = 0; ot < 4; ++ot) {
      // B operand: lane<16 -> N=ot*16+l16, K=m0..m0+15 (contiguous in hT);
      //            lane>=16 -> K=m0+16..m0+31
      const _Float16* bp =
          hT + (size_t)(hd * OUT_DIM + ot * 16 + l16) * NN + m0 + 16 * half;
      V16HU B;
      B.q[0] = *(const uint4*)(bp + 0);
      B.q[1] = *(const uint4*)(bp + 8);
      acc[ot] = __builtin_amdgcn_wmma_f32_16x16x32_f16(
          false, A, false, B.h, (short)0, acc[ot], false, false);
    }
  }

  // row sum: merge lane pair, then invert once
  S += __shfl_xor(S, 16, 32);
  const float invS = (S > 0.f) ? (1.f / S) : 0.f;

  // ----- Epilogue: normalize per row and store out[n][hd*64 + o] -----
  // acc VGPR r holds row r (lanes 0-15) / row r+8 (lanes 16-31); invS for row
  // j is resident (identically) at lanes j and j+16 -> srcLane = r + 8*half.
#pragma unroll
  for (int r = 0; r < 8; ++r) {
    const float sc   = __shfl(invS, r + 8 * half, 32);
    const int   orow = tile * 16 + r + 8 * half;
    float* op = out + (size_t)orow * (NHEAD * OUT_DIM) + hd * OUT_DIM;
#pragma unroll
    for (int ot = 0; ot < 4; ++ot) op[ot * 16 + l16] = acc[ot][r] * sc;
  }
}

// ---------------------------------------------------------------------------
extern "C" void kernel_launch(void* const* d_in, const int* in_sizes, int n_in,
                              void* d_out, int out_size, void* d_ws, size_t ws_size,
                              hipStream_t stream) {
  const float* x       = (const float*)d_in[0];   // [4096,128]
  const float* adj     = (const float*)d_in[1];   // [4096,4096]
  const float* W       = (const float*)d_in[2];   // [4,128,64]
  const float* a_self  = (const float*)d_in[3];   // [4,64]
  const float* a_neigh = (const float*)d_in[4];   // [4,64]
  float* out = (float*)d_out;                      // [4096,256]

  char* ws = (char*)d_ws;
  _Float16* hT   = (_Float16*)ws;                                   // 2 MB
  float*    sslf = (float*)(ws + (size_t)NHEAD * OUT_DIM * NN * 2); // 64 KB
  float*    sngh = sslf + NHEAD * NN;                               // 64 KB
  _Float16* Wp   = (_Float16*)(sngh + NHEAD * NN);                  // 64 KB

  gat_wprep <<<8, 256, 0, stream>>>(W, Wp);
  gat_hproj <<<NHEAD * (NN / 16), 32,  0, stream>>>(x, Wp, hT);
  gat_scores<<<NHEAD * (NN / 256), 256, 0, stream>>>(hT, a_self, a_neigh, sslf, sngh);
  gat_attn  <<<NN / 16, 32 * NHEAD, 0, stream>>>(adj, hT, sslf, sngh, out);
}
